// MultiHeadAttention_23545010717525
// MI455X (gfx1250) — compile-verified, matrix operand order checked by reading, not run on hardware
//
#include <hip/hip_runtime.h>
#include <hip/hip_bf16.h>

#define NB     4
#define SEQL   2048
#define NH     16
#define HD     32
#define DIM    512
#define ROWS   (NB*SEQL)        // 8192
#define CHUNK  16
#define NCH    (SEQL/CHUNK)     // 128 chunks per (n,h)
#define NHEADS (NB*NH)          // 64
#define NCHUNKS (NHEADS*NCH)    // 8192 total chunk tiles

typedef __attribute__((ext_vector_type(16))) __bf16 v16bf;
typedef __attribute__((ext_vector_type(8)))  float  v8f;
typedef __attribute__((ext_vector_type(4)))  unsigned int u32x4;
typedef __attribute__((ext_vector_type(8)))  int i32x8;
typedef __attribute__((ext_vector_type(4)))  int i32x4;

#if __has_builtin(__builtin_amdgcn_tensor_load_to_lds) && __has_builtin(__builtin_amdgcn_s_wait_tensorcnt)
#define USE_TDM 1
#else
#define USE_TDM 0
#endif

__device__ __forceinline__ v8f wmma_bf16(v16bf a, v16bf b, v8f c) {
  // D = A(16x32) * B(32x16) + C, f32 accumulate
  return __builtin_amdgcn_wmma_f32_16x16x32_bf16(false, a, false, b, (short)0, c, false, false);
}

// ---- Tensor Data Mover: stage one contiguous row of `nelem` bf16 into LDS ---
// Builds a D# per CDNA5 ISA: group0 {count=1, lds_addr,
// global_addr, type=2}; group1 {data_size=2B, tensor_dim0=nelem,
// tensor_dim1=1, tile_dim0=nelem, tile_dim1=1, dim0_stride=nelem}.
// One TDM op per wave (EXEC ignored), tracked by TENSORcnt.
__device__ __forceinline__ void tdm_load_row(__bf16* ldst, const __bf16* gsrc, int nelem) {
#if USE_TDM
  unsigned lds_off = __builtin_amdgcn_readfirstlane((unsigned)(size_t)(void*)ldst);
  unsigned long long ga = (unsigned long long)(size_t)(const void*)gsrc;
  unsigned ga_lo = __builtin_amdgcn_readfirstlane((unsigned)ga);
  unsigned ga_hi = __builtin_amdgcn_readfirstlane((unsigned)(ga >> 32));
  u32x4 g0;
  g0[0] = 1u;                                      // count=1 (valid user D#)
  g0[1] = lds_off;                                 // lds_addr (bytes)
  g0[2] = ga_lo;                                   // global_addr[95:64]
  g0[3] = (ga_hi & 0x01FFFFFFu) | (2u << 30);      // global_addr hi, type=2
  i32x8 g1;
  g1[0] = 0x00010000;                              // data_size=1 => 2 bytes
  g1[1] = (int)((nelem & 0xFFFF) << 16);           // tensor_dim0 lo16
  g1[2] = (int)((nelem >> 16) | (1 << 16));        // tensor_dim0 hi16, tensor_dim1=1
  g1[3] = (int)((nelem & 0xFFFF) << 16);           // tile_dim0 = nelem
  g1[4] = 1;                                       // tile_dim1 = 1
  g1[5] = (int)nelem;                              // tensor_dim0_stride lo32
  g1[6] = 0;
  g1[7] = 0;
  i32x4 gz = {0, 0, 0, 0};
#if __clang_major__ >= 23
  i32x8 gz8 = {0, 0, 0, 0, 0, 0, 0, 0};
  __builtin_amdgcn_tensor_load_to_lds(g0, g1, gz, gz, gz8, 0);
#else
  __builtin_amdgcn_tensor_load_to_lds(g0, g1, gz, gz, 0);
#endif
#else
  int lane = threadIdx.x & 31;
  int per = nelem >> 5;
  for (int j = 0; j < per; ++j) ldst[lane * per + j] = gsrc[lane * per + j];
#endif
}

__device__ __forceinline__ void tdm_wait() {
#if USE_TDM
  __builtin_amdgcn_s_wait_tensorcnt(0);
#endif
}

// A-fragment of a row-major 16x32 bf16 tile (lane&15 = M row; lane half picks
// the 8-wide K sub-runs per ISA 7.12.2: lanes<16 hold K {0-7,16-23}, lanes>=16
// hold K {8-15,24-31}). Also serves as the B operand of a *transposed* row-major
// matrix (B(kk,n) = M[n][kk]), since A/B lane-K patterns mirror each other.
__device__ __forceinline__ v16bf frag_a_rowmajor(const __bf16* base, int ld, int k0) {
  int lane = threadIdx.x & 31;
  int m = lane & 15, hh = (lane >> 4) & 1;
  const __bf16* p0 = base + m * ld + k0 + 8 * hh;
  const __bf16* p1 = p0 + 16;
  v16bf a;
#pragma unroll
  for (int j = 0; j < 8; ++j) a[j] = p0[j];
#pragma unroll
  for (int j = 0; j < 8; ++j) a[8 + j] = p1[j];
  return a;
}

// B-fragment (32x16) from a row-major KxN matrix, column block nbase.
// kreal < 32 zero-pads the missing K rows (used for K-dim = 16 cases).
__device__ __forceinline__ v16bf frag_b_rowmajor(const __bf16* base, int ld, int nbase, int kreal) {
  int lane = threadIdx.x & 31;
  int n = lane & 15, hh = (lane >> 4) & 1;
  v16bf b;
#pragma unroll
  for (int j = 0; j < 8; ++j) {
    int kk = 8 * hh + j;
    b[j] = (kk < kreal) ? base[kk * ld + nbase + n] : (__bf16)0.f;
  }
#pragma unroll
  for (int j = 0; j < 8; ++j) {
    int kk = 16 + 8 * hh + j;
    b[8 + j] = (kk < kreal) ? base[kk * ld + nbase + n] : (__bf16)0.f;
  }
  return b;
}

// ---------------- K0: weights f32 -> bf16 ----------------
__global__ void __launch_bounds__(256) k_cvt_w(const float* __restrict__ Wq,
                                               const float* __restrict__ Wk,
                                               const float* __restrict__ Wv,
                                               __bf16* __restrict__ wsW) {
  int idx = blockIdx.x * 256 + threadIdx.x;
  if (idx >= 3 * DIM * DIM) return;
  int mat = idx / (DIM * DIM);
  int rem = idx - mat * (DIM * DIM);
  const float* src = (mat == 0) ? Wq : (mat == 1 ? Wk : Wv);
  wsW[idx] = (__bf16)src[rem];
}

// ---------------- K1: projection GEMM + fused softmax, pack (n,h,l,d) bf16 ----
// grid = (ROWS/16, 3 outputs, 4 head-groups); 128 threads = 4 waves.
// Each wave owns one head's 16x32 output block (two WMMA N-tiles).
__global__ void __launch_bounds__(128) k_proj(const float* __restrict__ query,
                                              const float* __restrict__ key,
                                              const __bf16* __restrict__ wsW,
                                              __bf16* __restrict__ wsQ,
                                              __bf16* __restrict__ wsK,
                                              __bf16* __restrict__ wsV) {
  __shared__ __align__(16) __bf16 sX[16 * 520];   // 16 rows x 512 (+8 pad) bf16
  int rt = blockIdx.x, outp = blockIdx.y, hg = blockIdx.z;
  const float* X = (outp == 0) ? query : key;
  int rowbase = rt * 16;
  for (int i = threadIdx.x; i < 16 * 512; i += 128) {
    int r = i >> 9, d = i & 511;
    sX[r * 520 + d] = (__bf16)X[(size_t)(rowbase + r) * DIM + d];
  }
  __syncthreads();

  int wave = threadIdx.x >> 5, lane = threadIdx.x & 31;
  int head = hg * 4 + wave;
  int c0 = head * HD;
  const __bf16* W = wsW + (size_t)outp * DIM * DIM;

  v8f acc0 = {};
  v8f acc1 = {};
  for (int k0 = 0; k0 < DIM; k0 += 32) {
    v16bf a  = frag_a_rowmajor(sX, 520, k0);
    // y = x @ W^T  =>  B(kk,n) = W[c0+n][kk]: transposed-row-major pattern
    v16bf b0 = frag_a_rowmajor(W + (size_t)c0 * DIM, DIM, k0);
    v16bf b1 = frag_a_rowmajor(W + (size_t)(c0 + 16) * DIM, DIM, k0);
    acc0 = wmma_bf16(a, b0, acc0);
    acc1 = wmma_bf16(a, b1, acc1);
    if (k0 + 32 < DIM)
      __builtin_prefetch(W + (size_t)(c0 + (lane & 15)) * DIM + k0 + 32, 0, 1);
  }

  if (outp < 2) {
    // softmax over the head's 32 columns: row m lives in one 16-lane half at
    // VGPR index r, cols 0-15 in acc0[r], 16-31 in acc1[r].
#pragma unroll
    for (int r = 0; r < 8; ++r) {
      float mx = fmaxf(acc0[r], acc1[r]);
#pragma unroll
      for (int off = 8; off >= 1; off >>= 1) mx = fmaxf(mx, __shfl_xor(mx, off, 16));
      float e0 = __expf(acc0[r] - mx), e1 = __expf(acc1[r] - mx);
      float s = e0 + e1;
#pragma unroll
      for (int off = 8; off >= 1; off >>= 1) s += __shfl_xor(s, off, 16);
      float inv = 1.f / s;
      acc0[r] = e0 * inv;
      acc1[r] = e1 * inv;
    }
  }

  __bf16* dst = (outp == 0) ? wsQ : (outp == 1 ? wsK : wsV);
  int hh = lane >> 4, nn = lane & 15;
#pragma unroll
  for (int r = 0; r < 8; ++r) {
    int m = r + 8 * hh;
    int row = rowbase + m;
    int bN = row >> 11;           // / SEQL
    int l = row & (SEQL - 1);
    size_t base = (((size_t)(bN * NH + head)) * SEQL + l) * HD;
    dst[base + nn]      = (__bf16)acc0[r];
    dst[base + 16 + nn] = (__bf16)acc1[r];
  }
}

// ---------------- K2a: per-chunk KVc = Kc^T Vc (32x32 f32), all parallel ----
__global__ void __launch_bounds__(128) k_chunk_kv(const __bf16* __restrict__ wsK,
                                                  const __bf16* __restrict__ wsV,
                                                  float* __restrict__ wsKV) {
  __shared__ __align__(16) __bf16 sK[4][512];
  __shared__ __align__(16) __bf16 sV[4][512];
  int wave = threadIdx.x >> 5, lane = threadIdx.x & 31;
  int cid = blockIdx.x * 4 + wave;
  size_t cbase = (size_t)cid * 512;   // chunks are contiguous in (n,h,l,d)
  tdm_load_row(sK[wave], wsK + cbase, 512);   // async DMA -> LDS (TENSORcnt)
  tdm_load_row(sV[wave], wsV + cbase, 512);
  tdm_wait();
  // A = Kc^T (32x16 padded to K=32), B = Vc (16x32 rows, K padded)
  v16bf a0 = frag_b_rowmajor(sK[wave], 32, 0, 16);
  v16bf a1 = frag_b_rowmajor(sK[wave], 32, 16, 16);
  v16bf b0 = frag_b_rowmajor(sV[wave], 32, 0, 16);
  v16bf b1 = frag_b_rowmajor(sV[wave], 32, 16, 16);
  v8f kv00 = {}, kv01 = {}, kv10 = {}, kv11 = {};
  kv00 = wmma_bf16(a0, b0, kv00);
  kv01 = wmma_bf16(a0, b1, kv01);
  kv10 = wmma_bf16(a1, b0, kv10);
  kv11 = wmma_bf16(a1, b1, kv11);

  int hh = lane >> 4, nn = lane & 15;
  float* kvout = wsKV + (size_t)cid * 1024;
#pragma unroll
  for (int r = 0; r < 8; ++r) {
    int d0 = r + 8 * hh;
    kvout[d0 * 32 + nn]             = kv00[r];
    kvout[d0 * 32 + 16 + nn]        = kv01[r];
    kvout[(d0 + 16) * 32 + nn]      = kv10[r];
    kvout[(d0 + 16) * 32 + 16 + nn] = kv11[r];
  }
}

// ---------------- K2b: exclusive prefix over chunks, emit bf16 states --------
__global__ void __launch_bounds__(256) k_prefix(const float* __restrict__ wsKV,
                                                __bf16* __restrict__ wsS) {
  int nh = blockIdx.x;
  int t = threadIdx.x;      // 256 threads x 4 elems = 1024 state elements
  float a0 = 0.f, a1 = 0.f, a2 = 0.f, a3 = 0.f;
  for (int c = 0; c < NCH; ++c) {
    size_t base = ((size_t)nh * NCH + c) * 1024 + t * 4;
    wsS[base + 0] = (__bf16)a0;
    wsS[base + 1] = (__bf16)a1;
    wsS[base + 2] = (__bf16)a2;
    wsS[base + 3] = (__bf16)a3;
    float4 v = *(const float4*)(wsKV + base);
    a0 += v.x; a1 += v.y; a2 += v.z; a3 += v.w;
  }
}

// ---------------- K2c: out = Qc @ S_excl + tril(Qc Kc^T) Vc ------------------
__global__ void __launch_bounds__(128) k_attn_out(const __bf16* __restrict__ wsQ,
                                                  const __bf16* __restrict__ wsK,
                                                  const __bf16* __restrict__ wsV,
                                                  const __bf16* __restrict__ wsS,
                                                  float* __restrict__ out) {
  __shared__ __align__(16) __bf16 sQ[4][512];
  __shared__ __align__(16) __bf16 sK[4][512];
  __shared__ __align__(16) __bf16 sV[4][512];
  __shared__ __align__(16) __bf16 sS[4][1024];
  __shared__ __align__(16) float  sScr[4][256];
  int wave = threadIdx.x >> 5, lane = threadIdx.x & 31;
  int cid = blockIdx.x * 4 + wave;
  int nh = cid >> 7, c = cid & (NCH - 1);
  int nb = nh >> 4, head = nh & 15;
  size_t cbase = (size_t)cid * 512;
  size_t sbase = (size_t)cid * 1024;
  tdm_load_row(sQ[wave], wsQ + cbase, 512);   // 4 async TDM tiles per wave
  tdm_load_row(sK[wave], wsK + cbase, 512);
  tdm_load_row(sV[wave], wsV + cbase, 512);
  tdm_load_row(sS[wave], wsS + sbase, 1024);
  tdm_wait();

  v16bf aq = frag_a_rowmajor(sQ[wave], 32, 0);

  // inter-chunk contribution: Qc(16x32) @ S(32x32)
  v16bf bs0 = frag_b_rowmajor(sS[wave], 32, 0, 32);
  v16bf bs1 = frag_b_rowmajor(sS[wave], 32, 16, 32);
  v8f o0 = {}, o1 = {};
  o0 = wmma_bf16(aq, bs0, o0);
  o1 = wmma_bf16(aq, bs1, o1);

  // intra-chunk causal scores: Qc @ Kc^T (16x16)
  v16bf bk = frag_a_rowmajor(sK[wave], 32, 0);  // B(kk,n) = K[n][kk]
  v8f sc = {};
  sc = wmma_bf16(aq, bk, sc);

  int hh = lane >> 4, nn = lane & 15;
#pragma unroll
  for (int r = 0; r < 8; ++r) {
    int m = r + 8 * hh;                         // query row in chunk
    sScr[wave][m * 16 + nn] = (nn <= m) ? sc[r] : 0.f;  // causal (inclusive)
  }
  asm volatile("s_wait_dscnt 0" ::: "memory");  // in-wave LDS transpose hazard

  // repack masked scores as bf16 A-fragment (16x32, K rows 16..31 zero)
  v16bf a2;
  {
    int m = nn;
#pragma unroll
    for (int j = 0; j < 8; ++j) a2[j] = (__bf16)sScr[wave][m * 16 + 8 * hh + j];
#pragma unroll
    for (int j = 0; j < 8; ++j) a2[8 + j] = (__bf16)0.f;
  }
  v16bf bv0 = frag_b_rowmajor(sV[wave], 32, 0, 16);
  v16bf bv1 = frag_b_rowmajor(sV[wave], 32, 16, 16);
  o0 = wmma_bf16(a2, bv0, o0);
  o1 = wmma_bf16(a2, bv1, o1);

  // store to (N, L, H*D) f32
#pragma unroll
  for (int r = 0; r < 8; ++r) {
    int m = r + 8 * hh;
    int l = c * CHUNK + m;
    size_t ob = ((size_t)(nb * SEQL + l)) * DIM + head * HD;
    out[ob + nn]      = o0[r];
    out[ob + 16 + nn] = o1[r];
  }
}

extern "C" void kernel_launch(void* const* d_in, const int* in_sizes, int n_in,
                              void* d_out, int out_size, void* d_ws, size_t ws_size,
                              hipStream_t stream) {
  const float* query = (const float*)d_in[0];
  const float* key   = (const float*)d_in[1];
  const float* Wq    = (const float*)d_in[2];
  const float* Wk    = (const float*)d_in[3];
  const float* Wv    = (const float*)d_in[4];
  float* out = (float*)d_out;

  char* ws = (char*)d_ws;
  const size_t NHLD = (size_t)NHEADS * SEQL * HD;            // 4,194,304
  __bf16* wsW = (__bf16*)ws;                                 // 786,432 bf16
  __bf16* wsQ = wsW + 3 * DIM * DIM;
  __bf16* wsK = wsQ + NHLD;
  __bf16* wsV = wsK + NHLD;
  size_t bfBytes = ((size_t)3 * DIM * DIM + 3 * NHLD) * 2;   // 26,738,688 B
  float*  wsKV = (float*)(ws + bfBytes);                     // 8,388,608 f32
  __bf16* wsS  = (__bf16*)(ws + bfBytes + (size_t)NCHUNKS * 1024 * 4);

  k_cvt_w<<<3072, 256, 0, stream>>>(Wq, Wk, Wv, wsW);
  dim3 g1(ROWS / 16, 3, 4);
  k_proj<<<g1, 128, 0, stream>>>(query, key, wsW, wsQ, wsK, wsV);
  k_chunk_kv<<<NCHUNKS / 4, 128, 0, stream>>>(wsK, wsV, wsKV);
  k_prefix<<<NHEADS, 256, 0, stream>>>(wsKV, wsS);
  k_attn_out<<<NCHUNKS / 4, 128, 0, stream>>>(wsQ, wsK, wsV, wsS, out);
}
